// SpGraphAttentionLayer_11184094839120
// MI455X (gfx1250) — compile-verified, matrix operand order checked by reading, not run on hardware
//
#include <hip/hip_runtime.h>
#include <cstdint>
#include <cstddef>

// ---------------- problem constants (fixed by reference) ----------------
#define NNODES 50000
#define NEDGES 800000
#define F_IN   512
#define F_OUT  256
#define LRELU_ALPHA 0.2f

// WMMA vector types (gfx1250, wave32)
typedef __attribute__((ext_vector_type(16))) __bf16 v16bf;
typedef __attribute__((ext_vector_type(8)))  __bf16 v8bf;
typedef __attribute__((ext_vector_type(8)))  float  v8f;

static __device__ __forceinline__ unsigned bf16_rne(float f) {
    union { float f; unsigned u; } c; c.f = f;
    return (c.u + 0x7FFFu + ((c.u >> 16) & 1u)) >> 16;   // bf16 bits in [15:0]
}
static __device__ __forceinline__ unsigned pack2_bf16(float lo, float hi) {
    return bf16_rne(lo) | (bf16_rne(hi) << 16);
}
static __device__ __forceinline__ __bf16 f2bf(float f) {
    unsigned short s = (unsigned short)bf16_rne(f);
    __bf16 b;
    __builtin_memcpy(&b, &s, sizeof(b));
    return b;
}

// Load one 16x32 bf16 fragment row-slice: two contiguous 16B LDS reads.
static __device__ __forceinline__ v16bf load_frag(const __bf16* __restrict__ row, int half) {
    v8bf lo = *(const v8bf*)(row + half * 8);
    v8bf hi = *(const v8bf*)(row + 16 + half * 8);
    return __builtin_shufflevector(lo, hi, 0, 1, 2, 3, 4, 5, 6, 7,
                                   8, 9, 10, 11, 12, 13, 14, 15);
}

// Async global->LDS 16B copy (CDNA5 Async DMA path; tracked by ASYNCcnt).
#define ASYNC_CP_B128(ldsaddr, gaddr)                                          \
    asm volatile("global_load_async_to_lds_b128 %0, %1, off"                   \
                 :: "v"(ldsaddr), "v"(gaddr) : "memory")

// ---------------- kernel 0: zero int buffer ----------------
__global__ void zero_i32_kernel(int* __restrict__ p, int n) {
    int i = blockIdx.x * blockDim.x + threadIdx.x;
    if (i < n) p[i] = 0;
}

// ---------------- kernel 1a: x (fp32) -> xb (bf16, same row-major layout) ----------------
__global__ void cvt_x_kernel(const float* __restrict__ x, __bf16* __restrict__ xb) {
    size_t base = ((size_t)blockIdx.x * 256 + threadIdx.x) * 8;
    if (base >= (size_t)NNODES * F_IN) return;
    const float4* xp = (const float4*)(x + base);
    float4 f0 = xp[0];
    float4 f1 = xp[1];
    uint4 pk;
    pk.x = pack2_bf16(f0.x, f0.y);
    pk.y = pack2_bf16(f0.z, f0.w);
    pk.z = pack2_bf16(f1.x, f1.y);
    pk.w = pack2_bf16(f1.z, f1.w);
    *(uint4*)(xb + base) = pk;
}

// ---------------- kernel 1b: W (fp32, [K][N]) -> Wt (bf16, [N][K]) ----------------
__global__ void cvt_wt_kernel(const float* __restrict__ W, __bf16* __restrict__ Wt) {
    int i = blockIdx.x * blockDim.x + threadIdx.x;
    if (i >= F_IN * F_OUT) return;
    int n = i & (F_OUT - 1);
    int k = i >> 8;                       // F_OUT == 256
    Wt[(size_t)n * F_IN + k] = f2bf(W[i]);
}

// ---------------- kernel 2: h = xb @ W via bf16 WMMA, async double-buffered LDS ----------------
// Block tile 128x64, BK=32, 8 waves: wm = wave>>1 (4 strips of 32 rows),
// wn = wave&1 (2 strips of 32 cols). Each wave: 2x2 WMMA tiles (32x32 C).
// Staging: 3 global_load_async_to_lds_b128 per thread per tile, double-buffered.
#define BM 128
#define BN 64
#define BK 32
#define KPAD 8   // row stride 40 bf16 = 80 B -> 16 distinct bank groups across rows
__global__ __launch_bounds__(256) void gemm_wmma_kernel(const __bf16* __restrict__ xb,
                                                        const __bf16* __restrict__ Wt,
                                                        float* __restrict__ h) {
    __shared__ __bf16 As[2][BM][BK + KPAD];   // rows of xb
    __shared__ __bf16 Bs[2][BN][BK + KPAD];   // column-major W: Bs[.][n][k]

    const int tid  = threadIdx.x;
    const int wave = tid >> 5;
    const int lane = tid & 31;
    const int wm   = wave >> 1;            // 0..3
    const int wn   = wave & 1;             // 0..1
    const int half = lane >> 4;
    const int l15  = lane & 15;

    const int block_n = blockIdx.x * BN;   // N fastest -> same-M blocks adjacent (L2 reuse of x)
    const int block_m = blockIdx.y * BM;

    v8f acc00 = {0.f,0.f,0.f,0.f,0.f,0.f,0.f,0.f};
    v8f acc01 = {0.f,0.f,0.f,0.f,0.f,0.f,0.f,0.f};
    v8f acc10 = {0.f,0.f,0.f,0.f,0.f,0.f,0.f,0.f};
    v8f acc11 = {0.f,0.f,0.f,0.f,0.f,0.f,0.f,0.f};

    const int srow = tid >> 2;             // 0..63
    const int sseg = (tid & 3) * 8;        // k segment: 0,8,16,24

    // This thread's three staging slices (global side, bf16 elements)
    int gr0 = block_m + srow;        gr0 = (gr0 < NNODES) ? gr0 : (NNODES - 1);
    int gr1 = block_m + srow + 64;   gr1 = (gr1 < NNODES) ? gr1 : (NNODES - 1);
    const __bf16* ga0 = xb + (size_t)gr0 * F_IN + sseg;
    const __bf16* ga1 = xb + (size_t)gr1 * F_IN + sseg;
    const __bf16* gb  = Wt + (size_t)(block_n + srow) * F_IN + sseg;

    // LDS destinations (32-bit LDS offsets) per buffer
    unsigned lA0[2], lA1[2], lB[2];
    lA0[0] = (unsigned)(uintptr_t)&As[0][srow][sseg];
    lA0[1] = (unsigned)(uintptr_t)&As[1][srow][sseg];
    lA1[0] = (unsigned)(uintptr_t)&As[0][srow + 64][sseg];
    lA1[1] = (unsigned)(uintptr_t)&As[1][srow + 64][sseg];
    lB[0]  = (unsigned)(uintptr_t)&Bs[0][srow][sseg];
    lB[1]  = (unsigned)(uintptr_t)&Bs[1][srow][sseg];

    // prologue: async-stage tile 0 into buffer 0
    ASYNC_CP_B128(lA0[0], (unsigned long long)(uintptr_t)ga0);
    ASYNC_CP_B128(lA1[0], (unsigned long long)(uintptr_t)ga1);
    ASYNC_CP_B128(lB[0],  (unsigned long long)(uintptr_t)gb);

    for (int k0 = 0; k0 < F_IN; k0 += BK) {
        const int buf  = (k0 >> 5) & 1;
        const bool hasNext = (k0 + BK < F_IN);
        if (hasNext) {
            // async-stage next tile into the other buffer while this one computes
            ASYNC_CP_B128(lA0[buf ^ 1], (unsigned long long)(uintptr_t)(ga0 + k0 + BK));
            ASYNC_CP_B128(lA1[buf ^ 1], (unsigned long long)(uintptr_t)(ga1 + k0 + BK));
            ASYNC_CP_B128(lB[buf ^ 1],  (unsigned long long)(uintptr_t)(gb  + k0 + BK));
            asm volatile("s_wait_asynccnt 0x3" ::: "memory");  // current tile landed
        } else {
            asm volatile("s_wait_asynccnt 0x0" ::: "memory");
        }
        __syncthreads();   // all waves' async writes to `buf` visible

        // fragments: each 2x ds_load_b128
        v16bf a0 = load_frag(&As[buf][wm * 32 + l15][0], half);
        v16bf a1 = load_frag(&As[buf][wm * 32 + 16 + l15][0], half);
        v16bf b0 = load_frag(&Bs[buf][wn * 32 + l15][0], half);
        v16bf b1 = load_frag(&Bs[buf][wn * 32 + 16 + l15][0], half);

        acc00 = __builtin_amdgcn_wmma_f32_16x16x32_bf16(false, a0, false, b0,
                                                        (short)0, acc00, false, false);
        acc01 = __builtin_amdgcn_wmma_f32_16x16x32_bf16(false, a0, false, b1,
                                                        (short)0, acc01, false, false);
        acc10 = __builtin_amdgcn_wmma_f32_16x16x32_bf16(false, a1, false, b0,
                                                        (short)0, acc10, false, false);
        acc11 = __builtin_amdgcn_wmma_f32_16x16x32_bf16(false, a1, false, b1,
                                                        (short)0, acc11, false, false);
        __syncthreads();   // done reading `buf`; safe to async-overwrite two tiles ahead
    }

    // C/D layout: VGPR i -> row M = i + 8*half, col N = l15
    const int gn = block_n + wn * 32 + l15;
    #pragma unroll
    for (int i = 0; i < 8; ++i) {
        int gm0 = block_m + wm * 32 + 8 * half + i;
        int gm1 = gm0 + 16;
        if (gm0 < NNODES) {
            h[(size_t)gm0 * F_OUT + gn]      = acc00[i];
            h[(size_t)gm0 * F_OUT + gn + 16] = acc01[i];
        }
        if (gm1 < NNODES) {
            h[(size_t)gm1 * F_OUT + gn]      = acc10[i];
            h[(size_t)gm1 * F_OUT + gn + 16] = acc11[i];
        }
    }
}

// ---------------- kernel 3: alpha_src/alpha_dst = h @ a1 / h @ a2 ----------------
__global__ __launch_bounds__(256) void alpha_kernel(const float* __restrict__ h,
                                                    const float* __restrict__ a,
                                                    float* __restrict__ asrc,
                                                    float* __restrict__ adst) {
    int wave = threadIdx.x >> 5, lane = threadIdx.x & 31;
    int node = blockIdx.x * 8 + wave;
    if (node >= NNODES) return;
    const float* hr = h + (size_t)node * F_OUT;
    float s0 = 0.f, s1 = 0.f;
    #pragma unroll
    for (int j = 0; j < F_OUT / 32; ++j) {
        int c = lane + 32 * j;
        float v = hr[c];
        s0 += v * a[c];
        s1 += v * a[F_OUT + c];
    }
    #pragma unroll
    for (int off = 16; off > 0; off >>= 1) {
        s0 += __shfl_down(s0, off, 32);
        s1 += __shfl_down(s1, off, 32);
    }
    if (lane == 0) { asrc[node] = s0; adst[node] = s1; }
}

// ---------------- kernel 4: per-edge weight + out-degree count ----------------
__global__ void edge_weight_kernel(const int* __restrict__ src, const int* __restrict__ dst,
                                   const float* __restrict__ asrc, const float* __restrict__ adst,
                                   float* __restrict__ ew, int* __restrict__ count) {
    int e = blockIdx.x * blockDim.x + threadIdx.x;
    if (e >= NEDGES) return;
    int s = src[e];
    float sc = asrc[s] + adst[dst[e]];
    float lr = (sc > 0.f) ? sc : LRELU_ALPHA * sc;
    ew[e] = __expf(-lr);
    atomicAdd(&count[s], 1);
}

// ---------------- kernel 5: exclusive scan of counts -> CSR starts ----------------
#define SCAN_CH 49   // ceil(50000 / 1024)
__global__ __launch_bounds__(1024) void scan_kernel(const int* __restrict__ count,
                                                    int* __restrict__ start,
                                                    int* __restrict__ cursor) {
    __shared__ int sums[1024];
    int t = threadIdx.x;
    int beg = t * SCAN_CH;
    int local = 0;
    for (int i = 0; i < SCAN_CH; ++i) {
        int idx = beg + i;
        if (idx < NNODES) local += count[idx];
    }
    sums[t] = local;
    __syncthreads();
    for (int off = 1; off < 1024; off <<= 1) {
        int v = (t >= off) ? sums[t - off] : 0;
        __syncthreads();
        sums[t] += v;
        __syncthreads();
    }
    int excl = sums[t] - local;
    for (int i = 0; i < SCAN_CH; ++i) {
        int idx = beg + i;
        if (idx < NNODES) {
            start[idx]  = excl;
            cursor[idx] = excl;
            excl += count[idx];
        }
    }
}

// ---------------- kernel 6: bucket edges by src (counting-sort scatter) ----------------
__global__ void scatter_kernel(const int* __restrict__ src, int* __restrict__ cursor,
                               int* __restrict__ perm) {
    int e = blockIdx.x * blockDim.x + threadIdx.x;
    if (e >= NEDGES) return;
    int pos = atomicAdd(&cursor[src[e]], 1);
    perm[pos] = e;
}

// ---------------- kernel 7: per-node aggregation, rownorm, ELU ----------------
__global__ __launch_bounds__(256) void aggregate_kernel(const float* __restrict__ h,
                                                        const int* __restrict__ dst,
                                                        const float* __restrict__ ew,
                                                        const int* __restrict__ perm,
                                                        const int* __restrict__ start,
                                                        const int* __restrict__ cursor,
                                                        float* __restrict__ out) {
    int wave = threadIdx.x >> 5, lane = threadIdx.x & 31;
    int node = blockIdx.x * 8 + wave;
    if (node >= NNODES) return;
    int b  = start[node];
    int en = cursor[node];   // after scatter, cursor == start + degree
    float acc[8] = {0.f, 0.f, 0.f, 0.f, 0.f, 0.f, 0.f, 0.f};
    float rs = 0.f;
    for (int i = b; i < en; ++i) {
        int e = perm[i];
        float w = ew[e];
        const float* hr = h + (size_t)dst[e] * F_OUT;
        rs += w;
        #pragma unroll
        for (int j = 0; j < 8; ++j) acc[j] += w * hr[lane + 32 * j];
    }
    float inv = 1.f / rs;    // rs > 0 guaranteed (self-loops)
    float* orow = out + (size_t)node * F_OUT;
    #pragma unroll
    for (int j = 0; j < 8; ++j) {
        float v = acc[j] * inv;
        orow[lane + 32 * j] = (v > 0.f) ? v : (__expf(v) - 1.f);
    }
}

// ---------------- host: launch pipeline ----------------
extern "C" void kernel_launch(void* const* d_in, const int* in_sizes, int n_in,
                              void* d_out, int out_size, void* d_ws, size_t ws_size,
                              hipStream_t stream) {
    const float* x  = (const float*)d_in[0];
    const int*   ei = (const int*)d_in[1];   // [2, E]: row0 = src, row1 = dst
    const float* W  = (const float*)d_in[2];
    const float* a  = (const float*)d_in[3];
    float* out = (float*)d_out;

    const int* src = ei;
    const int* dst = ei + NEDGES;

    // workspace layout (all section sizes are multiples of 16 B)
    char* p = (char*)d_ws;
    float*  h      = (float*)p;          p += (size_t)NNODES * F_OUT * sizeof(float);
    __bf16* xb     = (__bf16*)p;         p += (size_t)NNODES * F_IN * sizeof(__bf16);
    __bf16* Wt     = (__bf16*)p;         p += (size_t)F_IN * F_OUT * sizeof(__bf16);
    float*  asrc   = (float*)p;          p += (size_t)NNODES * sizeof(float);
    float*  adst   = (float*)p;          p += (size_t)NNODES * sizeof(float);
    float*  ew     = (float*)p;          p += (size_t)NEDGES * sizeof(float);
    int*    count  = (int*)p;            p += (size_t)NNODES * sizeof(int);
    int*    start  = (int*)p;            p += (size_t)NNODES * sizeof(int);
    int*    cursor = (int*)p;            p += (size_t)NNODES * sizeof(int);
    int*    perm   = (int*)p;            p += (size_t)NEDGES * sizeof(int);

    zero_i32_kernel<<<(NNODES + 255) / 256, 256, 0, stream>>>(count, NNODES);
    cvt_x_kernel<<<(int)(((size_t)NNODES * F_IN / 8 + 255) / 256), 256, 0, stream>>>(x, xb);
    cvt_wt_kernel<<<(F_IN * F_OUT + 255) / 256, 256, 0, stream>>>(W, Wt);

    dim3 ggrid(F_OUT / BN, (NNODES + BM - 1) / BM);   // (4, 391)
    gemm_wmma_kernel<<<ggrid, 256, 0, stream>>>(xb, Wt, h);

    alpha_kernel<<<(NNODES + 7) / 8, 256, 0, stream>>>(h, a, asrc, adst);
    edge_weight_kernel<<<(NEDGES + 255) / 256, 256, 0, stream>>>(src, dst, asrc, adst, ew, count);
    scan_kernel<<<1, 1024, 0, stream>>>(count, start, cursor);
    scatter_kernel<<<(NEDGES + 255) / 256, 256, 0, stream>>>(src, cursor, perm);
    aggregate_kernel<<<(NNODES + 7) / 8, 256, 0, stream>>>(h, dst, ew, perm, start, cursor, out);
}
